// EMAQuantizer_38663295599460
// MI455X (gfx1250) — matrix operand order repaired and checked by reading, not compiled, not run on hardware
//
#include <hip/hip_runtime.h>
#include <hip/hip_bf16.h>

// ---------------------------------------------------------------------------
// EMA vector-quantizer for MI455X (gfx1250, wave32, WMMA bf16)
// ---------------------------------------------------------------------------

typedef __attribute__((ext_vector_type(16))) __bf16 v16bf;
typedef __attribute__((ext_vector_type(8)))  __bf16 v8bf;
typedef __attribute__((ext_vector_type(8)))  float  v8f;

#define DEVI static __device__ __forceinline__

// ---- problem constants ----
#define N_TOK   16384      // 8*2048
#define E_DIM   256
#define LATENT  768
#define N_E     8192
#define DECAYF  0.99f
#define ONEMD   0.01f
#define EPSF    1e-5f
#define EXPIRE_THRESH 0.1f   // DEAD_THRESH * RATIO = 0.05*2
#define RESET_VAL     0.12f  // RESET_CS * RATIO = 0.06*2

// ---- workspace layout (bytes) ----
constexpr size_t OFF_WINB  = 0;                        // 256*768*2
constexpr size_t OFF_WOUTB = OFF_WINB  + 393216;       // 768*256*2
constexpr size_t OFF_EMBB  = OFF_WOUTB + 393216;       // 8192*256*2
constexpr size_t OFF_XN    = OFF_EMBB  + 4194304;      // 16384*256*4
constexpr size_t OFF_XNB   = OFF_XN    + 16777216;     // 16384*256*2
constexpr size_t OFF_ZQB   = OFF_XNB   + 8388608;      // 16384*256*2
constexpr size_t OFF_ROWZ  = OFF_ZQB   + 8388608;      // 16384*4
constexpr size_t OFF_ROWS  = OFF_ROWZ  + 65536;        // 16384*4
constexpr size_t OFF_IND   = OFF_ROWS  + 65536;        // 16384*4
constexpr size_t OFF_ESUM  = OFF_IND   + 65536;        // 8192*256*4  (zeroed region start)
constexpr size_t OFF_BINS  = OFF_ESUM  + 8388608;      // 8192*4
constexpr size_t OFF_AP    = OFF_BINS  + 32768;        // 8192*4
constexpr size_t OFF_SCAL  = OFF_AP    + 32768;        // [0]=total, [1]=commit_acc
constexpr size_t OFF_CNEW  = OFF_SCAL  + 64;           // 8192*4
constexpr size_t OFF_EXP   = OFF_CNEW  + 32768;        // 8192*4
constexpr size_t OFF_SIDX  = OFF_EXP   + 32768;        // 8192*4
constexpr int    ZERO_FLOATS = (8388608 + 32768 + 32768 + 64) / 4;

// ---- output layout (float elements, reference tuple order) ----
constexpr size_t O_ZQ   = 0;                           // 16384*768
constexpr size_t O_INDF = O_ZQ   + 12582912;           // 16384
constexpr size_t O_LOSS = O_INDF + 16384;              // 1
constexpr size_t O_EMB  = O_LOSS + 1;                  // 8192*256
constexpr size_t O_AVG  = O_EMB  + 2097152;            // 8192*256
constexpr size_t O_CS   = O_AVG  + 2097152;            // 8192
constexpr size_t O_ETM  = O_CS   + 8192;               // 1
constexpr size_t O_COM  = O_ETM  + 1;                  // 1

// ---------------------------------------------------------------------------
// WMMA fragment helpers (CDNA5 16x16x32 bf16 layouts, CDNA5 ISA)
// ---------------------------------------------------------------------------
DEVI v8f zero8() {
  v8f r;
#pragma unroll
  for (int i = 0; i < 8; ++i) r[i] = 0.f;
  return r;
}

DEVI v16bf combine8(v8bf lo, v8bf hi) {
  v16bf r;
#pragma unroll
  for (int j = 0; j < 8; ++j) { r[j] = lo[j]; r[j + 8] = hi[j]; }
  return r;
}

// A fragment 16x32 bf16: lanes 0-15 row M=lane hold K(kk+0..7, kk+16..23),
// lanes 16-31 row M=lane-16 hold K(kk+8..15, kk+24..31). Row-major source.
DEVI v16bf load_a_bf16(const __bf16* A, int lda, int row0, int kk, int lane) {
  const int m  = lane & 15;
  const int ko = (lane & 16) ? 8 : 0;
  const __bf16* p = A + (size_t)(row0 + m) * lda + (kk + ko);
  v8bf lo = *reinterpret_cast<const v8bf*>(p);
  v8bf hi = *reinterpret_cast<const v8bf*>(p + 16);
  return combine8(lo, hi);
}

// Same fragment, converting from fp32 row-major on the fly.
DEVI v16bf load_a_f32(const float* A, int lda, int row0, int kk, int lane) {
  const int m  = lane & 15;
  const int ko = (lane & 16) ? 8 : 0;
  const float* p = A + (size_t)(row0 + m) * lda + (kk + ko);
  v16bf r;
#pragma unroll
  for (int j = 0; j < 8; ++j) { r[j] = (__bf16)p[j]; r[j + 8] = (__bf16)p[j + 16]; }
  return r;
}

// B fragment 32x16 bf16: lanes 0-15 col N=lane hold K(kk+0..7, kk+16..23),
// lanes 16-31 col N=lane-16 hold K(kk+8..15, kk+24..31). Source is row-major
// [Ncols][K] (i.e. embed / torch-Linear weight layout), so per-lane loads
// are two contiguous 16-byte runs.
DEVI v16bf load_b_bf16(const __bf16* B, int ldb, int col0, int kk, int lane) {
  const int n  = lane & 15;
  const int kb = (lane & 16) ? 8 : 0;
  const __bf16* p = B + (size_t)(col0 + n) * ldb + (kk + kb);
  v8bf lo = *reinterpret_cast<const v8bf*>(p);
  v8bf hi = *reinterpret_cast<const v8bf*>(p + 16);
  return combine8(lo, hi);
}

DEVI v8f wmma_bf16(v16bf a, v16bf b, v8f c) {
  return __builtin_amdgcn_wmma_f32_16x16x32_bf16(false, a, false, b, (short)0, c,
                                                 false, false);
}

// C/D element i maps to (row = row0 + i + ((lane&16)?8:0), col = col0 + (lane&15))

// ---------------------------------------------------------------------------
// Utility kernels
// ---------------------------------------------------------------------------
__global__ __launch_bounds__(256) void k_zero(float* p, int n) {
  int i = blockIdx.x * 256 + threadIdx.x;
  if (i < n) p[i] = 0.f;
}

__global__ __launch_bounds__(256) void k_cvt_bf16(const float* __restrict__ s,
                                                  __bf16* __restrict__ d, int n) {
  int i = blockIdx.x * 256 + threadIdx.x;
  if (i < n) d[i] = (__bf16)s[i];
}

// ---------------------------------------------------------------------------
// K1: x_proj = x @ W_in^T + b_in, then row-l2norm -> xn (f32 + bf16)
// grid 1024 (16 rows each), block 512 = 16 waves, one 16x16 N-tile per wave
// ---------------------------------------------------------------------------
__global__ __launch_bounds__(512) void k_proj_in(const float* __restrict__ x,
                                                 const __bf16* __restrict__ WinB,
                                                 const float* __restrict__ b_in,
                                                 float* __restrict__ xn,
                                                 __bf16* __restrict__ xnb) {
  const int row0 = blockIdx.x * 16;
  const int lane = threadIdx.x & 31;
  const int wave = threadIdx.x >> 5;
  const int col0 = wave * 16;
  v8f c = zero8();
  for (int kk = 0; kk < LATENT; kk += 32) {
    v16bf a = load_a_f32(x, LATENT, row0, kk, lane);
    v16bf b = load_b_bf16(WinB, LATENT, col0, kk, lane);
    c = wmma_bf16(a, b, c);
  }
  const int ncol = lane & 15;
  const int roff = (lane & 16) ? 8 : 0;
  const float bias = b_in[col0 + ncol];
  float sq[8];
#pragma unroll
  for (int i = 0; i < 8; ++i) { float v = c[i] + bias; c[i] = v; sq[i] = v * v; }
#pragma unroll
  for (int i = 0; i < 8; ++i)
#pragma unroll
    for (int off = 8; off >= 1; off >>= 1) sq[i] += __shfl_xor(sq[i], off, 16);

  __shared__ float ssum[16][16];
  __shared__ float sscale[16];
  if (lane == 0 || lane == 16) {
#pragma unroll
    for (int i = 0; i < 8; ++i) ssum[wave][i + roff] = sq[i];
  }
  __syncthreads();
  if (wave == 0 && lane < 16) {
    float t = 0.f;
#pragma unroll
    for (int w = 0; w < 16; ++w) t += ssum[w][lane];
    sscale[lane] = rsqrtf(t);
  }
  __syncthreads();
#pragma unroll
  for (int i = 0; i < 8; ++i) {
    const int r = i + roff;
    const float v = c[i] * sscale[r];
    const size_t o = (size_t)(row0 + r) * E_DIM + col0 + ncol;
    xn[o] = v;
    xnb[o] = (__bf16)v;
  }
}

// ---------------------------------------------------------------------------
// K2 (pass 1): d = xn @ embed^T, fused argmin + Z=sum(e^{10d}) + S=sum(d e^{10d})
// grid 1024 row-blocks, block 256 = 8 waves; each wave sweeps cols stride 128.
// A-fragments for the full K=256 kept register-resident.
// ---------------------------------------------------------------------------
__global__ __launch_bounds__(256) void k_pass1(const __bf16* __restrict__ xnb,
                                               const __bf16* __restrict__ embb,
                                               float* __restrict__ rowZ,
                                               float* __restrict__ rowS,
                                               int* __restrict__ rowInd,
                                               float* __restrict__ indF) {
  const int row0 = blockIdx.x * 16;
  const int lane = threadIdx.x & 31;
  const int wave = threadIdx.x >> 5;
  v16bf afr[8];
#pragma unroll
  for (int k = 0; k < 8; ++k) afr[k] = load_a_bf16(xnb, E_DIM, row0, k * 32, lane);

  float zacc[8], sacc[8], mval[8];
  int midx[8];
#pragma unroll
  for (int i = 0; i < 8; ++i) { zacc[i] = 0.f; sacc[i] = 0.f; mval[i] = 3.4e38f; midx[i] = 0; }

  for (int c0 = wave * 16; c0 < N_E; c0 += 128) {
    if (c0 + 128 < N_E)  // hint next B tile into cache (global_prefetch)
      __builtin_prefetch(embb + (size_t)(c0 + 128) * E_DIM, 0, 1);
    v8f c = zero8();
#pragma unroll
    for (int k = 0; k < 8; ++k) {
      v16bf b = load_b_bf16(embb, E_DIM, c0, k * 32, lane);
      c = wmma_bf16(afr[k], b, c);
    }
    const int mycol = c0 + (lane & 15);
#pragma unroll
    for (int i = 0; i < 8; ++i) {
      const float d = c[i];
      if (d < mval[i]) { mval[i] = d; midx[i] = mycol; }
      const float e = __expf(10.f * d);
      zacc[i] += e;
      sacc[i] += d * e;
    }
  }
  // reduce across the 16 lanes sharing each row (halves stay separate)
#pragma unroll
  for (int i = 0; i < 8; ++i) {
#pragma unroll
    for (int off = 8; off >= 1; off >>= 1) {
      zacc[i] += __shfl_xor(zacc[i], off, 16);
      sacc[i] += __shfl_xor(sacc[i], off, 16);
      const float ov = __shfl_xor(mval[i], off, 16);
      const int   oi = __shfl_xor(midx[i], off, 16);
      if (ov < mval[i] || (ov == mval[i] && oi < midx[i])) { mval[i] = ov; midx[i] = oi; }
    }
  }
  __shared__ float Zl[8][16], Sl[8][16], Mv[8][16];
  __shared__ int   Mi[8][16];
  if (lane == 0 || lane == 16) {
    const int roff = (lane == 16) ? 8 : 0;
#pragma unroll
    for (int i = 0; i < 8; ++i) {
      Zl[wave][i + roff] = zacc[i];
      Sl[wave][i + roff] = sacc[i];
      Mv[wave][i + roff] = mval[i];
      Mi[wave][i + roff] = midx[i];
    }
  }
  __syncthreads();
  if (wave == 0 && lane < 16) {
    float Z = 0.f, S = 0.f, mv = 3.4e38f;
    int mi = 0;
#pragma unroll
    for (int w = 0; w < 8; ++w) {
      Z += Zl[w][lane];
      S += Sl[w][lane];
      const float v = Mv[w][lane];
      const int   ix = Mi[w][lane];
      if (v < mv || (v == mv && ix < mi)) { mv = v; mi = ix; }
    }
    rowZ[row0 + lane]   = Z;
    rowS[row0 + lane]   = S;
    rowInd[row0 + lane] = mi;
    indF[row0 + lane]   = (float)mi;
  }
}

// ---------------------------------------------------------------------------
// K3 (pass 2): ap_c = sum_n exp(10 d_nc) / Z_n  (column-stationary recompute)
// grid 128, block 128 = 4 waves * 16 cols; B-fragments register-resident.
// ---------------------------------------------------------------------------
__global__ __launch_bounds__(128) void k_pass2(const __bf16* __restrict__ xnb,
                                               const __bf16* __restrict__ embb,
                                               const float* __restrict__ rowZ,
                                               float* __restrict__ ap) {
  const int lane = threadIdx.x & 31;
  const int wave = threadIdx.x >> 5;
  const int col0 = blockIdx.x * 64 + wave * 16;
  v16bf bfr[8];
#pragma unroll
  for (int k = 0; k < 8; ++k) bfr[k] = load_b_bf16(embb, E_DIM, col0, k * 32, lane);

  float acc = 0.f;
  for (int r0 = 0; r0 < N_TOK; r0 += 16) {
    v8f c = zero8();
#pragma unroll
    for (int k = 0; k < 8; ++k) {
      v16bf a = load_a_bf16(xnb, E_DIM, r0, k * 32, lane);
      c = wmma_bf16(a, bfr[k], c);
    }
    const int rb = r0 + ((lane & 16) ? 8 : 0);
#pragma unroll
    for (int i = 0; i < 8; ++i) acc += __expf(10.f * c[i]) / rowZ[rb + i];
  }
  acc += __shfl_xor(acc, 16, 32);  // combine the two row-halves of each column
  if (lane < 16) ap[col0 + lane] = acc;
}

// ---------------------------------------------------------------------------
// K4: gather z_q=embed[ind], scatter embed_sum += xn, bins histogram,
//     commit-loss partial. grid 16384 (1 row), block 256 (= E_DIM).
// ---------------------------------------------------------------------------
__global__ __launch_bounds__(256) void k_gather(const int* __restrict__ rowInd,
                                                const float* __restrict__ embed,
                                                const float* __restrict__ xn,
                                                __bf16* __restrict__ zqb,
                                                float* __restrict__ embed_sum,
                                                float* __restrict__ bins,
                                                float* __restrict__ commit_acc) {
  const int row = blockIdx.x;
  const int e   = threadIdx.x;
  const int ind = rowInd[row];
  const float zq = embed[(size_t)ind * E_DIM + e];
  const float xv = xn[(size_t)row * E_DIM + e];
  zqb[(size_t)row * E_DIM + e] = (__bf16)zq;
  atomicAdd(&embed_sum[(size_t)ind * E_DIM + e], xv);
  const float d = zq - xv;
  __shared__ float red[256];
  red[e] = d * d;
  __syncthreads();
  for (int s = 128; s > 0; s >>= 1) {
    if (e < s) red[e] += red[e + s];
    __syncthreads();
  }
  if (e == 0) {
    atomicAdd(commit_acc, red[0]);
    atomicAdd(&bins[ind], 1.0f);
  }
}

// ---------------------------------------------------------------------------
// K5: z_q_out = z_q @ W_out^T + b_out. grid (1024, 6), block 256 = 8 waves.
// ---------------------------------------------------------------------------
__global__ __launch_bounds__(256) void k_out_gemm(const __bf16* __restrict__ zqb,
                                                  const __bf16* __restrict__ WoutB,
                                                  const float* __restrict__ b_out,
                                                  float* __restrict__ out) {
  const int row0 = blockIdx.x * 16;
  const int lane = threadIdx.x & 31;
  const int wave = threadIdx.x >> 5;
  const int col0 = blockIdx.y * 128 + wave * 16;
  v8f c = zero8();
  for (int kk = 0; kk < E_DIM; kk += 32) {
    v16bf a = load_a_bf16(zqb, E_DIM, row0, kk, lane);
    v16bf b = load_b_bf16(WoutB, E_DIM, col0, kk, lane);
    c = wmma_bf16(a, b, c);
  }
  const int ncol = lane & 15;
  const int roff = (lane & 16) ? 8 : 0;
  const float bias = b_out[col0 + ncol];
#pragma unroll
  for (int i = 0; i < 8; ++i)
    out[(size_t)(row0 + i + roff) * LATENT + col0 + ncol] = c[i] + bias;
}

// ---------------------------------------------------------------------------
// K6: cluster EMA + expired flags + total reduction. grid 32, block 256.
// ---------------------------------------------------------------------------
__global__ __launch_bounds__(256) void k_cluster(const float* __restrict__ cluster_size,
                                                 const float* __restrict__ bins,
                                                 float* __restrict__ cluster_new,
                                                 int* __restrict__ expired,
                                                 float* __restrict__ total) {
  const int i = blockIdx.x * 256 + threadIdx.x;
  const float c = cluster_size[i] * DECAYF + bins[i] * ONEMD;
  cluster_new[i] = c;
  expired[i] = (c < EXPIRE_THRESH) ? 1 : 0;
  __shared__ float red[256];
  red[threadIdx.x] = c;
  __syncthreads();
  for (int s = 128; s > 0; s >>= 1) {
    if (threadIdx.x < s) red[threadIdx.x] += red[threadIdx.x + s];
    __syncthreads();
  }
  if (threadIdx.x == 0) atomicAdd(total, red[0]);
}

// ---------------------------------------------------------------------------
// K7: inclusive prefix-sum over expired[8192] -> sample indices. 1 block.
// ---------------------------------------------------------------------------
__global__ __launch_bounds__(1024) void k_scan(const int* __restrict__ expired,
                                               int* __restrict__ sidx) {
  const int t = threadIdx.x;
  const int base = t * 8;
  int loc[8];
  int s = 0;
#pragma unroll
  for (int j = 0; j < 8; ++j) { s += expired[base + j]; loc[j] = s; }
  __shared__ int ls[1024];
  ls[t] = s;
  __syncthreads();
  for (int off = 1; off < 1024; off <<= 1) {
    int v = (t >= off) ? ls[t - off] : 0;
    __syncthreads();
    ls[t] += v;
    __syncthreads();
  }
  const int pre = t ? ls[t - 1] : 0;
#pragma unroll
  for (int j = 0; j < 8; ++j) {
    int cum = pre + loc[j];
    int idx = cum - 1;
    if (idx < 0) idx = 0;
    sidx[base + j] = idx & (N_TOK - 1);  // % 16384
  }
}

// ---------------------------------------------------------------------------
// K8: laplace smoothing, embed_norm, dead-code replacement; write final
// embed / embed_avg / cluster outputs. grid 8192, block 256.
// ---------------------------------------------------------------------------
__global__ __launch_bounds__(256) void k_final(const float* __restrict__ embed_avg,
                                               const float* __restrict__ embed_sum,
                                               const float* __restrict__ cluster_new,
                                               const int* __restrict__ expired,
                                               const int* __restrict__ sidx,
                                               const float* __restrict__ xn,
                                               const float* __restrict__ total_p,
                                               float* __restrict__ emb_out,
                                               float* __restrict__ avg_out,
                                               float* __restrict__ cs_out) {
  const int c = blockIdx.x;
  const int e = threadIdx.x;
  const float cn  = cluster_new[c];
  const float avg = embed_avg[(size_t)c * E_DIM + e] * DECAYF +
                    embed_sum[(size_t)c * E_DIM + e] * ONEMD;
  const float total = *total_p;
  const float cs = (cn + EPSF) / (total + (float)N_E * EPSF) * total;
  const float en = avg / cs;
  const int ex = expired[c];
  const float samp = xn[(size_t)sidx[c] * E_DIM + e];
  emb_out[(size_t)c * E_DIM + e] = ex ? samp : en;
  avg_out[(size_t)c * E_DIM + e] = ex ? samp * RESET_VAL : avg;
  if (e == 0) cs_out[c] = ex ? RESET_VAL : cn;
}

// ---------------------------------------------------------------------------
// K9: scalar losses. 1 block.
//   entropy_to_max = -sum(ap log ap), ap = apacc/n
//   entropy_to_min = mean(log Z - 10 S/Z)
//   loss = commit + sum(ap log ap)   (diversity_loss = -entropy_to_max)
// ---------------------------------------------------------------------------
__global__ __launch_bounds__(256) void k_entropy(const float* __restrict__ ap,
                                                 const float* __restrict__ rowZ,
                                                 const float* __restrict__ rowS,
                                                 const float* __restrict__ commit_acc,
                                                 float* __restrict__ out_loss,
                                                 float* __restrict__ out_etm,
                                                 float* __restrict__ out_commit) {
  __shared__ float red[256];
  const int t = threadIdx.x;
  float h = 0.f;
  for (int i = t; i < N_E; i += 256) {
    const float a = ap[i] * (1.f / (float)N_TOK);
    if (a > 0.f) h += a * __logf(a);
  }
  red[t] = h;
  __syncthreads();
  for (int s = 128; s > 0; s >>= 1) {
    if (t < s) red[t] += red[t + s];
    __syncthreads();
  }
  const float hsum = red[0];  // = sum(ap log ap) = diversity_loss
  __syncthreads();
  float em = 0.f;
  for (int i = t; i < N_TOK; i += 256) {
    const float Z = rowZ[i];
    em += __logf(Z) - 10.f * rowS[i] / Z;
  }
  red[t] = em;
  __syncthreads();
  for (int s = 128; s > 0; s >>= 1) {
    if (t < s) red[t] += red[t + s];
    __syncthreads();
  }
  if (t == 0) {
    const float commit = commit_acc[0] * (1.f / ((float)N_TOK * (float)E_DIM));
    out_commit[0] = commit;
    out_etm[0]    = red[0] * (1.f / (float)N_TOK);
    out_loss[0]   = commit + hsum;
  }
}

// ---------------------------------------------------------------------------
extern "C" void kernel_launch(void* const* d_in, const int* in_sizes, int n_in,
                              void* d_out, int out_size, void* d_ws, size_t ws_size,
                              hipStream_t stream) {
  (void)in_sizes; (void)n_in; (void)out_size; (void)ws_size;
  const float* x      = (const float*)d_in[0];
  const float* W_in   = (const float*)d_in[1];
  const float* b_in   = (const float*)d_in[2];
  const float* W_out  = (const float*)d_in[3];
  const float* b_out  = (const float*)d_in[4];
  const float* embed  = (const float*)d_in[5];
  const float* e_avg  = (const float*)d_in[6];
  const float* c_size = (const float*)d_in[7];

  float* out = (float*)d_out;
  char*  ws  = (char*)d_ws;
  __bf16* WinB  = (__bf16*)(ws + OFF_WINB);
  __bf16* WoutB = (__bf16*)(ws + OFF_WOUTB);
  __bf16* EmbB  = (__bf16*)(ws + OFF_EMBB);
  float*  xn    = (float*)(ws + OFF_XN);
  __bf16* xnb   = (__bf16*)(ws + OFF_XNB);
  __bf16* zqb   = (__bf16*)(ws + OFF_ZQB);
  float*  rowZ  = (float*)(ws + OFF_ROWZ);
  float*  rowS  = (float*)(ws + OFF_ROWS);
  int*    rInd  = (int*)(ws + OFF_IND);
  float*  esum  = (float*)(ws + OFF_ESUM);
  float*  bins  = (float*)(ws + OFF_BINS);
  float*  ap    = (float*)(ws + OFF_AP);
  float*  total = (float*)(ws + OFF_SCAL);
  float*  commit= (float*)(ws + OFF_SCAL) + 1;
  float*  cnew  = (float*)(ws + OFF_CNEW);
  int*    expd  = (int*)(ws + OFF_EXP);
  int*    sidx  = (int*)(ws + OFF_SIDX);

  // 0) zero accumulators (embed_sum, bins, ap, scalars are contiguous)
  k_zero<<<(ZERO_FLOATS + 255) / 256, 256, 0, stream>>>((float*)(ws + OFF_ESUM), ZERO_FLOATS);
  // bf16 weight/codebook conversions
  k_cvt_bf16<<<(E_DIM * LATENT + 255) / 256, 256, 0, stream>>>(W_in, WinB, E_DIM * LATENT);
  k_cvt_bf16<<<(LATENT * E_DIM + 255) / 256, 256, 0, stream>>>(W_out, WoutB, LATENT * E_DIM);
  k_cvt_bf16<<<(N_E * E_DIM + 255) / 256, 256, 0, stream>>>(embed, EmbB, N_E * E_DIM);
  // 1) project in + l2norm
  k_proj_in<<<N_TOK / 16, 512, 0, stream>>>(x, WinB, b_in, xn, xnb);
  // 2) pass1: argmin / Z / S  (writes embed_ind floats straight to output)
  k_pass1<<<N_TOK / 16, 256, 0, stream>>>(xnb, EmbB, rowZ, rowS, rInd, out + O_INDF);
  // 3) pass2: ap column accumulation
  k_pass2<<<N_E / 64, 128, 0, stream>>>(xnb, EmbB, rowZ, ap);
  // 4) gather/scatter + commit partials
  k_gather<<<N_TOK, 256, 0, stream>>>(rInd, embed, xn, zqb, esum, bins, commit);
  // 5) output projection
  k_out_gemm<<<dim3(N_TOK / 16, LATENT / 128), 256, 0, stream>>>(zqb, WoutB, b_out, out + O_ZQ);
  // 6) EMA cluster + total
  k_cluster<<<N_E / 256, 256, 0, stream>>>(c_size, bins, cnew, expd, total);
  // 7) dead-code prefix scan
  k_scan<<<1, 1024, 0, stream>>>(expd, sidx);
  // 8) final embed / avg / cluster outputs
  k_final<<<N_E, 256, 0, stream>>>(e_avg, esum, cnew, expd, sidx, xn, total,
                                   out + O_EMB, out + O_AVG, out + O_CS);
  // 9) scalar losses
  k_entropy<<<1, 256, 0, stream>>>(ap, rowZ, rowS, commit,
                                   out + O_LOSS, out + O_ETM, out + O_COM);
}
